// CrossAttention_5583457484928
// MI455X (gfx1250) — matrix operand order repaired and checked by reading, not compiled, not run on hardware
//
#include <hip/hip_runtime.h>
#include <hip/hip_bf16.h>

// ---------------- CDNA5 WMMA types ----------------
typedef __attribute__((ext_vector_type(16))) __bf16 v16bf;
typedef __attribute__((ext_vector_type(8)))  float  v8f;

struct __align__(16) U4 { unsigned x, y, z, w; };

union FragAB {
    v16bf v;
    U4    u[2];
    unsigned short h[16];
};

#define WMMA_BF16(A, B, C) \
    __builtin_amdgcn_wmma_f32_16x16x32_bf16(false, (A), false, (B), (short)0, (C), false, false)

// ---------------- constants ----------------
#define BATCH   2
#define SEQ     2048
#define DMODEL  1024
#define HEADS   16
#define DHEAD   64
#define NPOS    1025          // 2*512+1
#define SCALE_F 0.125f        // 64^-0.5

// ---------------- helpers ----------------
__device__ __forceinline__ unsigned short f2bf(float f) {
    union { float f; unsigned u; } v; v.f = f;
    unsigned r = v.u + 0x7FFFu + ((v.u >> 16) & 1u);
    return (unsigned short)(r >> 16);
}
__device__ __forceinline__ float bf2f(unsigned short b) {
    union { unsigned u; float f; } v; v.u = ((unsigned)b) << 16;
    return v.f;
}

// A-frag: rows M = lane&15, K run [kb+g*16, +16) contiguous, row-major src
__device__ __forceinline__ void load_a(FragAB& f, const unsigned short* A, int ld,
                                       int r0, int kb, int lane) {
    int M = lane & 15, g = lane >> 4;
    const unsigned short* row = A + (size_t)(r0 + M) * ld;
    f.u[0] = *reinterpret_cast<const U4*>(row + kb + g * 16);
    f.u[1] = *reinterpret_cast<const U4*>(row + kb + g * 16 + 8);
}

// B-frag from BT[N][K] row-major: col n = lane&15, K run [kb+g*16, +16) contiguous
__device__ __forceinline__ void load_bt(FragAB& f, const unsigned short* BT, int ld,
                                        int n0, int kb, int lane) {
    int n = lane & 15, g = lane >> 4;
    const unsigned short* base = BT + (size_t)(n0 + n) * ld + kb + g * 16;
    f.u[0] = reinterpret_cast<const U4*>(base)[0];
    f.u[1] = reinterpret_cast<const U4*>(base)[1];
}

// ---------------- conversion kernels ----------------
__global__ void k_cvt_bf16(const float* __restrict__ src, unsigned short* __restrict__ dst, size_t n) {
    size_t i = (size_t)blockIdx.x * blockDim.x + threadIdx.x;
    if (i < n) dst[i] = f2bf(src[i]);
}

// src [K][N] f32 -> dst [N][K] bf16
__global__ void k_transpose_bf16(const float* __restrict__ src, unsigned short* __restrict__ dst,
                                 int K, int N) {
    size_t i = (size_t)blockIdx.x * blockDim.x + threadIdx.x;
    if (i >= (size_t)K * N) return;
    int c = (int)(i / N), n = (int)(i % N);
    dst[(size_t)n * K + c] = f2bf(src[i]);
}

// =====================================================================
// Register-blocked GEMM core: each wave computes a 32x64 output tile
// (2 M-frags x 4 N-frags -> 8 WMMAs per k-step from 12 B128 loads).
// =====================================================================
struct Acc2x4 { v8f a[2][4]; };

__device__ __forceinline__ void gemm_32x64(Acc2x4& acc,
                                           const unsigned short* __restrict__ A, int lda,
                                           const unsigned short* __restrict__ BT, int ldb,
                                           int r0, int n0, int K, int lane) {
#pragma unroll
    for (int mi = 0; mi < 2; ++mi)
#pragma unroll
        for (int ni = 0; ni < 4; ++ni)
            acc.a[mi][ni] = (v8f){0.f, 0.f, 0.f, 0.f, 0.f, 0.f, 0.f, 0.f};

    int M = lane & 15, g = lane >> 4, n = lane & 15;
    const unsigned short* arow0 = A + (size_t)(r0 + M) * lda;
    const unsigned short* arow1 = A + (size_t)(r0 + 16 + M) * lda;

    for (int k = 0; k < K; k += 32) {
        // prefetch next k-step (lowers to global_prefetch_b8)
        __builtin_prefetch(arow0 + k + 64, 0, 1);
        __builtin_prefetch(arow1 + k + 64, 0, 1);

        FragAB a0, a1;
        a0.u[0] = *reinterpret_cast<const U4*>(arow0 + k + g * 16);
        a0.u[1] = *reinterpret_cast<const U4*>(arow0 + k + g * 16 + 8);
        a1.u[0] = *reinterpret_cast<const U4*>(arow1 + k + g * 16);
        a1.u[1] = *reinterpret_cast<const U4*>(arow1 + k + g * 16 + 8);

#pragma unroll
        for (int ni = 0; ni < 4; ++ni) {
            FragAB b;
            const unsigned short* brow = BT + (size_t)(n0 + ni * 16 + n) * ldb + k + g * 16;
            __builtin_prefetch(brow + 64, 0, 1);
            b.u[0] = reinterpret_cast<const U4*>(brow)[0];
            b.u[1] = reinterpret_cast<const U4*>(brow)[1];
            acc.a[0][ni] = WMMA_BF16(a0.v, b.v, acc.a[0][ni]);
            acc.a[1][ni] = WMMA_BF16(a1.v, b.v, acc.a[1][ni]);
        }
    }
}

// ---------------- q = (x @ Wq) * SCALE, stored (b,h,i,c) bf16 ----------------
__global__ __launch_bounds__(32) void k_gemm_q(const unsigned short* __restrict__ xbf,
                                               const unsigned short* __restrict__ WqT,
                                               unsigned short* __restrict__ qws) {
    int nt = blockIdx.x & 15, mt = blockIdx.x >> 4;   // 16 col blocks (64), 128 row blocks (32)
    int lane = threadIdx.x;
    int r0 = mt * 32, n0 = nt * 64;
    Acc2x4 acc;
    gemm_32x64(acc, xbf, DMODEL, WqT, DMODEL, r0, n0, DMODEL, lane);

    int n = lane & 15, g = lane >> 4;
#pragma unroll
    for (int mi = 0; mi < 2; ++mi)
#pragma unroll
        for (int ni = 0; ni < 4; ++ni) {
            int col = n0 + ni * 16 + n, h = col >> 6, c = col & 63;
#pragma unroll
            for (int r = 0; r < 8; ++r) {
                int row = r0 + mi * 16 + r + 8 * g;
                int b_ = row >> 11, i = row & 2047;
                qws[((size_t)(b_ * HEADS + h) * SEQ + i) * DHEAD + c] =
                    f2bf(acc.a[mi][ni][r] * SCALE_F);
            }
        }
}

// ---------------- kv = x @ Wkv; k -> (b,h,j,c), v -> (b,h,c,j) bf16 ----------------
__global__ __launch_bounds__(32) void k_gemm_kv(const unsigned short* __restrict__ xbf,
                                                const unsigned short* __restrict__ WkvT,
                                                unsigned short* __restrict__ kws,
                                                unsigned short* __restrict__ vTws) {
    int nt = blockIdx.x & 31, mt = blockIdx.x >> 5;   // 32 col blocks (64), 128 row blocks (32)
    int lane = threadIdx.x;
    int r0 = mt * 32, n0 = nt * 64;
    Acc2x4 acc;
    gemm_32x64(acc, xbf, DMODEL, WkvT, DMODEL, r0, n0, DMODEL, lane);

    int n = lane & 15, g = lane >> 4;
    if (n0 < DMODEL) {                      // K half (64-col block never straddles 1024)
#pragma unroll
        for (int mi = 0; mi < 2; ++mi)
#pragma unroll
            for (int ni = 0; ni < 4; ++ni) {
                int col = n0 + ni * 16 + n, h = col >> 6, c = col & 63;
#pragma unroll
                for (int r = 0; r < 8; ++r) {
                    int row = r0 + mi * 16 + r + 8 * g;
                    int b_ = row >> 11, j = row & 2047;
                    kws[((size_t)(b_ * HEADS + h) * SEQ + j) * DHEAD + c] =
                        f2bf(acc.a[mi][ni][r]);
                }
            }
    } else {                                // V half, transposed (b,h,c,j)
#pragma unroll
        for (int mi = 0; mi < 2; ++mi)
#pragma unroll
            for (int ni = 0; ni < 4; ++ni) {
                int c2 = n0 + ni * 16 + n - DMODEL, h = c2 >> 6, c = c2 & 63;
#pragma unroll
                for (int r = 0; r < 8; ++r) {
                    int row = r0 + mi * 16 + r + 8 * g;
                    int b_ = row >> 11, j = row & 2047;
                    vTws[((size_t)(b_ * HEADS + h) * DHEAD + c) * SEQ + j] =
                        f2bf(acc.a[mi][ni][r]);
                }
            }
    }
}

// ---------------- fused attention: one wave per (b,h, 16-row tile) ----------------
__global__ __launch_bounds__(32) void k_attention(const unsigned short* __restrict__ qws,
                                                  const unsigned short* __restrict__ kws,
                                                  const unsigned short* __restrict__ vTws,
                                                  const unsigned short* __restrict__ relbf,
                                                  unsigned short* __restrict__ attnws) {
    int it = blockIdx.x & 127, bh = blockIdx.x >> 7;
    int b_ = bh >> 4, h = bh & 15;
    int i0 = it * 16;
    int lane = threadIdx.x, nloc = lane & 15, g = lane >> 4;

    __shared__ __align__(16) unsigned short qrel[16][1040];  // bf16, p in [0,1024]
    __shared__ __align__(16) unsigned short Plds[16][32];    // bf16 P tile

    const unsigned short* qb = qws  + (size_t)bh * SEQ * DHEAD;
    const unsigned short* kb = kws  + (size_t)bh * SEQ * DHEAD;
    const unsigned short* vb = vTws + (size_t)bh * DHEAD * SEQ;

    // q A-fragments (K = feature dim, 64 = two k-steps); SCALE already folded in
    FragAB qa0, qa1;
    load_a(qa0, qb, DHEAD, i0, 0, lane);
    load_a(qa1, qb, DHEAD, i0, 32, lane);

    // qrel[i_loc][p] = q_row . rel_emb[p]  (65 N-tiles of 16, tail clamped)
    for (int pt = 0; pt < 65; ++pt) {
        int p0 = pt * 16;
        int pr = p0 + nloc; if (pr > 1024) pr = 1024;
        FragAB rb0, rb1;
        const unsigned short* rrow = relbf + (size_t)pr * DHEAD;
        rb0.u[0] = reinterpret_cast<const U4*>(rrow + 0 + g * 16)[0];
        rb0.u[1] = reinterpret_cast<const U4*>(rrow + 0 + g * 16)[1];
        rb1.u[0] = reinterpret_cast<const U4*>(rrow + 32 + g * 16)[0];
        rb1.u[1] = reinterpret_cast<const U4*>(rrow + 32 + g * 16)[1];
        v8f acc = {0.f, 0.f, 0.f, 0.f, 0.f, 0.f, 0.f, 0.f};
        acc = WMMA_BF16(qa0.v, rb0.v, acc);
        acc = WMMA_BF16(qa1.v, rb1.v, acc);
        if (p0 + nloc <= 1024) {
#pragma unroll
            for (int r = 0; r < 8; ++r) qrel[r + 8 * g][p0 + nloc] = f2bf(acc[r]);
        }
    }
    __syncthreads();

    // online-softmax state
    float mrow[8], lsum[8];
    v8f Oacc[4];
#pragma unroll
    for (int r = 0; r < 8; ++r) { mrow[r] = -3.0e38f; lsum[r] = 0.f; }
#pragma unroll
    for (int dt = 0; dt < 4; ++dt)
        Oacc[dt] = (v8f){0.f, 0.f, 0.f, 0.f, 0.f, 0.f, 0.f, 0.f};

    for (int j0 = 0; j0 < SEQ; j0 += 32) {
        // two 16x16 score tiles (keys j0..j0+31)
        v8f s0 = {0.f, 0.f, 0.f, 0.f, 0.f, 0.f, 0.f, 0.f};
        v8f s1 = {0.f, 0.f, 0.f, 0.f, 0.f, 0.f, 0.f, 0.f};
        {
            FragAB kf;
            load_bt(kf, kb, DHEAD, j0, 0, lane);        s0 = WMMA_BF16(qa0.v, kf.v, s0);
            load_bt(kf, kb, DHEAD, j0, 32, lane);       s0 = WMMA_BF16(qa1.v, kf.v, s0);
            load_bt(kf, kb, DHEAD, j0 + 16, 0, lane);   s1 = WMMA_BF16(qa0.v, kf.v, s1);
            load_bt(kf, kb, DHEAD, j0 + 16, 32, lane);  s1 = WMMA_BF16(qa1.v, kf.v, s1);
        }
        // + relative position term from LDS, then online softmax
        float fac[8], mnew[8];
#pragma unroll
        for (int r = 0; r < 8; ++r) {
            int iloc = r + 8 * g, ig = i0 + iloc;
            int j_a = j0 + nloc,      j_b = j0 + 16 + nloc;
            int pa = ig - j_a + 512;  pa = pa < 0 ? 0 : (pa > 1024 ? 1024 : pa);
            int pb = ig - j_b + 512;  pb = pb < 0 ? 0 : (pb > 1024 ? 1024 : pb);
            s0[r] += bf2f(qrel[iloc][pa]);
            s1[r] += bf2f(qrel[iloc][pb]);
            float c = fmaxf(s0[r], s1[r]);
            c = fmaxf(c, __shfl_xor(c, 1));
            c = fmaxf(c, __shfl_xor(c, 2));
            c = fmaxf(c, __shfl_xor(c, 4));
            c = fmaxf(c, __shfl_xor(c, 8));
            mnew[r] = fmaxf(mrow[r], c);
            fac[r]  = __expf(mrow[r] - mnew[r]);
            mrow[r] = mnew[r];
            float p0v = __expf(s0[r] - mnew[r]);
            float p1v = __expf(s1[r] - mnew[r]);
            s0[r] = p0v; s1[r] = p1v;
            float rs = p0v + p1v;
            rs += __shfl_xor(rs, 1);
            rs += __shfl_xor(rs, 2);
            rs += __shfl_xor(rs, 4);
            rs += __shfl_xor(rs, 8);
            lsum[r] = lsum[r] * fac[r] + rs;
        }
#pragma unroll
        for (int dt = 0; dt < 4; ++dt)
#pragma unroll
            for (int r = 0; r < 8; ++r) Oacc[dt][r] *= fac[r];

        // P (C/D layout) -> LDS bf16 -> A-layout fragment
        __syncthreads();
#pragma unroll
        for (int r = 0; r < 8; ++r) {
            Plds[r + 8 * g][nloc]      = f2bf(s0[r]);
            Plds[r + 8 * g][16 + nloc] = f2bf(s1[r]);
        }
        __syncthreads();
        FragAB pa;
        pa.u[0] = *reinterpret_cast<const U4*>(&Plds[nloc][g * 16]);
        pa.u[1] = *reinterpret_cast<const U4*>(&Plds[nloc][g * 16 + 8]);

        // O += P @ V  (V pre-transposed: contiguous B-frags)
#pragma unroll
        for (int dt = 0; dt < 4; ++dt) {
            FragAB vf;
            const unsigned short* vrow = vb + (size_t)(dt * 16 + nloc) * SEQ + j0 + g * 16;
            vf.u[0] = reinterpret_cast<const U4*>(vrow)[0];
            vf.u[1] = reinterpret_cast<const U4*>(vrow)[1];
            Oacc[dt] = WMMA_BF16(pa.v, vf.v, Oacc[dt]);
        }
    }

    // normalize + write (b, i, h*d) bf16 for the output projection
#pragma unroll
    for (int r = 0; r < 8; ++r) {
        float inv = 1.0f / lsum[r];
        int row = i0 + r + 8 * g;
#pragma unroll
        for (int dt = 0; dt < 4; ++dt) {
            int col = h * DHEAD + dt * 16 + nloc;
            attnws[(size_t)(b_ * SEQ + row) * DMODEL + col] = f2bf(Oacc[dt][r] * inv);
        }
    }
}

// ---------------- out = attn @ Wo + bo (f32 output) ----------------
__global__ __launch_bounds__(32) void k_gemm_out(const unsigned short* __restrict__ attnbf,
                                                 const unsigned short* __restrict__ WoT,
                                                 const float* __restrict__ bo,
                                                 float* __restrict__ out) {
    int nt = blockIdx.x & 15, mt = blockIdx.x >> 4;   // 16 col blocks (64), 128 row blocks (32)
    int lane = threadIdx.x;
    int r0 = mt * 32, n0 = nt * 64;
    Acc2x4 acc;
    gemm_32x64(acc, attnbf, DMODEL, WoT, DMODEL, r0, n0, DMODEL, lane);

    int n = lane & 15, g = lane >> 4;
#pragma unroll
    for (int ni = 0; ni < 4; ++ni) {
        int col = n0 + ni * 16 + n;
        float bias = bo[col];
#pragma unroll
        for (int mi = 0; mi < 2; ++mi)
#pragma unroll
            for (int r = 0; r < 8; ++r) {
                int row = r0 + mi * 16 + r + 8 * g;
                out[(size_t)row * DMODEL + col] = acc.a[mi][ni][r] + bias;
            }
    }
}

// ---------------- host launcher ----------------
extern "C" void kernel_launch(void* const* d_in, const int* in_sizes, int n_in,
                              void* d_out, int out_size, void* d_ws, size_t ws_size,
                              hipStream_t stream) {
    const float* x    = (const float*)d_in[0];  // (2,2048,1024)
    const float* Wq   = (const float*)d_in[1];  // (1024,1024)
    const float* Wkv  = (const float*)d_in[2];  // (1024,2048)
    const float* Wo   = (const float*)d_in[3];  // (1024,1024)
    const float* bo   = (const float*)d_in[4];  // (1024,)
    const float* rel  = (const float*)d_in[5];  // (1025,64)
    float* out = (float*)d_out;

    char* ws = (char*)d_ws;
    size_t off = 0;
    auto alloc = [&](size_t bytes) { char* p = ws + off; off = (off + bytes + 255) & ~(size_t)255; return p; };

    unsigned short* xbf    = (unsigned short*)alloc((size_t)BATCH * SEQ * DMODEL * 2);
    unsigned short* WqT    = (unsigned short*)alloc((size_t)DMODEL * DMODEL * 2);
    unsigned short* WkvT   = (unsigned short*)alloc((size_t)2 * DMODEL * DMODEL * 2);
    unsigned short* WoT    = (unsigned short*)alloc((size_t)DMODEL * DMODEL * 2);
    unsigned short* relbf  = (unsigned short*)alloc((size_t)NPOS * DHEAD * 2);
    unsigned short* qws    = (unsigned short*)alloc((size_t)BATCH * HEADS * SEQ * DHEAD * 2);
    unsigned short* kws    = (unsigned short*)alloc((size_t)BATCH * HEADS * SEQ * DHEAD * 2);
    unsigned short* vTws   = (unsigned short*)alloc((size_t)BATCH * HEADS * SEQ * DHEAD * 2);
    unsigned short* attnws = (unsigned short*)alloc((size_t)BATCH * SEQ * DMODEL * 2);

    const size_t n_x   = (size_t)BATCH * SEQ * DMODEL;
    const size_t n_rel = (size_t)NPOS * DHEAD;

    k_cvt_bf16<<<(unsigned)((n_x + 255) / 256), 256, 0, stream>>>(x, xbf, n_x);
    k_transpose_bf16<<<(unsigned)((DMODEL * DMODEL + 255) / 256), 256, 0, stream>>>(Wq, WqT, DMODEL, DMODEL);
    k_transpose_bf16<<<(unsigned)((2 * DMODEL * DMODEL + 255) / 256), 256, 0, stream>>>(Wkv, WkvT, DMODEL, 2 * DMODEL);
    k_transpose_bf16<<<(unsigned)((DMODEL * DMODEL + 255) / 256), 256, 0, stream>>>(Wo, WoT, DMODEL, DMODEL);
    k_cvt_bf16<<<(unsigned)((n_rel + 255) / 256), 256, 0, stream>>>(rel, relbf, n_rel);

    k_gemm_q <<<128 * 16, 32, 0, stream>>>(xbf, WqT, qws);
    k_gemm_kv<<<128 * 32, 32, 0, stream>>>(xbf, WkvT, kws, vTws);
    k_attention<<<BATCH * HEADS * (SEQ / 16), 32, 0, stream>>>(qws, kws, vTws, relbf, attnws);
    k_gemm_out<<<128 * 16, 32, 0, stream>>>(attnws, WoT, bo, out);
}
